// CrossAttentionLayer_33973191311811
// MI455X (gfx1250) — compile-verified, matrix operand order checked by reading, not run on hardware
//
#include <hip/hip_runtime.h>

// CDNA5 / gfx1250, wave32. Fused LN + SwiGLU FF via v_wmma_f32_16x16x32_bf16.
// Attention path is computed exactly in closed form (softmax over constant
// logits is uniform -> attention output == broadcast v_row), so x@Wq is skipped.

typedef __attribute__((ext_vector_type(16))) __bf16 v16bf;
typedef __attribute__((ext_vector_type(8)))  float  v8f;
typedef __attribute__((ext_vector_type(4)))  unsigned int v4u;

union Frag16 { v16bf bf; v4u u[2]; };

#define B_      16
#define C_      256
#define HW_     4096
#define EDIM    512
#define MT      64      // tokens per block tile
#define XN_STR  264     // bf16 row stride (pad: 264*2=528B, m*132dw -> distinct banks)
#define ACT_STR 72      // bf16 row stride (144B rows)
#define ST_STR  68      // f32 staging row stride

__device__ __forceinline__ v8f wmma_bf16(const Frag16& a, const Frag16& b, v8f c) {
  return __builtin_amdgcn_wmma_f32_16x16x32_bf16(false, a.bf, false, b.bf,
                                                 (short)0, c, false, false);
}

// ---- kernel 1: transpose + f32->bf16 convert weights into workspace --------
__global__ void conv_w_kernel(const float* __restrict__ W1,   // (256,1024)
                              const float* __restrict__ W2,   // (512,256)
                              __bf16* __restrict__ w1t,       // (1024,256)
                              __bf16* __restrict__ w2t) {     // (256,512)
  int id = blockIdx.x * 256 + threadIdx.x;
  if (id < 1024 * 256) {
    int k = id & 255, n = id >> 8;
    w1t[id] = (__bf16)W1[k * 1024 + n];
  } else {
    int id2 = id - 1024 * 256;
    int k = id2 & 511, n = id2 >> 9;
    w2t[id2] = (__bf16)W2[k * 256 + n];
  }
}

// ---- kernel 2: closed-form attention row per batch -------------------------
// e = LN(extra); v_row = e @ Wkv[:,64:128]; attn_row = tile8(v_row) @ Wout.
__global__ __launch_bounds__(256) void attn_row_kernel(
    const float* __restrict__ extra, const float* __restrict__ g2,
    const float* __restrict__ b2, const float* __restrict__ wkv,
    const float* __restrict__ wout, float* __restrict__ attn) {
  __shared__ float e_s[EDIM];
  __shared__ float v_s[64];
  __shared__ float red[256];
  __shared__ float red2[256];
  const int b = blockIdx.x, t = threadIdx.x;
  float x0 = extra[b * EDIM + t], x1 = extra[b * EDIM + 256 + t];
  red[t] = x0 + x1; red2[t] = x0 * x0 + x1 * x1;
  __syncthreads();
  for (int s = 128; s > 0; s >>= 1) {
    if (t < s) { red[t] += red[t + s]; red2[t] += red2[t + s]; }
    __syncthreads();
  }
  float mu = red[0] * (1.f / 512.f);
  float var = red2[0] * (1.f / 512.f) - mu * mu;
  float rs = rsqrtf(var + 1e-5f);
  e_s[t]       = (x0 - mu) * rs * g2[t]       + b2[t];
  e_s[t + 256] = (x1 - mu) * rs * g2[t + 256] + b2[t + 256];
  __syncthreads();
  if (t < 64) {
    float s = 0.f;
    for (int i = 0; i < EDIM; ++i) s += e_s[i] * wkv[i * 128 + 64 + t];
    v_s[t] = s;
  }
  __syncthreads();
  float s = 0.f;
  for (int k = 0; k < EDIM; ++k) s += v_s[k & 63] * wout[k * 256 + t];
  attn[b * 256 + t] = s;
}

// ---- kernel 3: fused LN + SwiGLU FF (WMMA) + attn add + transposed store ---
__global__ __launch_bounds__(256, 1) void ff_fused_kernel(
    const float* __restrict__ x,        // (B, C, 4096) channel-major
    const float* __restrict__ g1, const float* __restrict__ b1,
    const __bf16* __restrict__ w1t,     // (1024, 256) bf16, N-major
    const __bf16* __restrict__ w2t,     // (256, 512) bf16, N-major
    const float* __restrict__ attn,     // (B, 256)
    float* __restrict__ out) {          // (B, C, 4096)
  __shared__ float   xstage[C_][ST_STR];      // f32 staging (reused for output)
  __shared__ __bf16  xn[MT][XN_STR];          // LN'd activations (64 x 256)
  __shared__ __bf16  act[MT][ACT_STR];        // silu(gate)*xh chunk (64 x 64)
  __shared__ float   psum[4][64], psq[4][64];
  __shared__ float   mu_s[64], rs_s[64];
  __shared__ float   attn_s[C_], g1s[C_], b1s[C_];

  const int t    = threadIdx.x;
  const int b    = blockIdx.x >> 6;
  const int hw0  = (blockIdx.x & 63) << 6;
  const int lane = t & 31;
  const int w    = t >> 5;            // wave 0..7
  const int lrow  = lane & 15;        // row/col within 16-lane half
  const int khalf = (lane >> 4) << 3; // K window offset (+8 for upper half)
  const int mrow  = (lane >> 4) << 3; // C-fragment M offset

  // ---- load x tile (coalesced along hw), preload LN params + attn row ----
  const float* xb = x + ((size_t)b * C_) * HW_ + hw0;
  attn_s[t & 255] = attn[b * C_ + (t & 255)];
  g1s[t & 255] = g1[t & 255];
  b1s[t & 255] = b1[t & 255];
  #pragma unroll 4
  for (int i = 0; i < 64; ++i) {
    int idx = t + 256 * i;            // 0..16383
    int c = idx >> 6, m = idx & 63;
    xstage[c][m] = xb[(size_t)c * HW_ + m];
  }
  __syncthreads();

  // ---- LayerNorm stats over C for each of 64 tokens ----
  {
    int m = t & 63, p = t >> 6;
    float s = 0.f, q = 0.f;
    #pragma unroll 8
    for (int c = p * 64; c < p * 64 + 64; ++c) {
      float v = xstage[c][m]; s += v; q += v * v;
    }
    psum[p][m] = s; psq[p][m] = q;
  }
  __syncthreads();
  if (t < 64) {
    float s = psum[0][t] + psum[1][t] + psum[2][t] + psum[3][t];
    float q = psq[0][t] + psq[1][t] + psq[2][t] + psq[3][t];
    float mu = s * (1.f / 256.f);
    float var = q * (1.f / 256.f) - mu * mu;
    mu_s[t] = mu; rs_s[t] = rsqrtf(var + 1e-5f);
  }
  __syncthreads();
  #pragma unroll 4
  for (int i = 0; i < 64; ++i) {
    int idx = t + 256 * i; int c = idx >> 6, m = idx & 63;
    float v = (xstage[c][m] - mu_s[m]) * rs_s[m] * g1s[c] + b1s[c];
    xn[m][c] = (__bf16)v;
  }
  __syncthreads();

  // ---- persistent output accumulators: wave owns m-tile (w>>1), 8 n-tiles ----
  const v8f z8 = {0.f, 0.f, 0.f, 0.f, 0.f, 0.f, 0.f, 0.f};
  v8f oacc[8];
  #pragma unroll
  for (int j = 0; j < 8; ++j) oacc[j] = z8;

  const int mt  = w >> 1;         // m-tile 0..3 (shared by phases 1 & 2)
  const int nt0 = (w & 1) * 2;    // phase-1 n-tile pair within 64-wide chunk
  const int ntb = (w & 1) * 8;    // phase-2 n-tile base over 256 output cols

  for (int chunk = 0; chunk < 8; ++chunk) {
    // ---- phase 1: Hx = Xn@W1[:,chunk], Hg = Xn@W1[:,512+chunk] ----
    v8f ax[2] = {z8, z8}, ag[2] = {z8, z8};
    for (int k0 = 0; k0 < 256; k0 += 32) {
      int kb = k0 + khalf;
      Frag16 a;
      const __bf16* ap = &xn[mt * 16 + lrow][kb];
      a.u[0] = *(const v4u*)ap;
      a.u[1] = *(const v4u*)(ap + 16);
      #pragma unroll
      for (int j = 0; j < 2; ++j) {
        int ncx = chunk * 64 + (nt0 + j) * 16 + lrow;
        const __bf16* bp = w1t + (size_t)ncx * 256 + kb;
        Frag16 bx; bx.u[0] = *(const v4u*)bp; bx.u[1] = *(const v4u*)(bp + 16);
        ax[j] = wmma_bf16(a, bx, ax[j]);
        const __bf16* gp = bp + (size_t)512 * 256;   // gate columns +512
        Frag16 bg; bg.u[0] = *(const v4u*)gp; bg.u[1] = *(const v4u*)(gp + 16);
        ag[j] = wmma_bf16(a, bg, ag[j]);
      }
    }
    // in-register SwiGLU (x and gate accumulators share lane mapping)
    #pragma unroll
    for (int j = 0; j < 2; ++j) {
      #pragma unroll
      for (int r = 0; r < 8; ++r) {
        float gz = ag[j][r];
        float val = ax[j][r] * (gz / (1.f + __expf(-gz)));
        act[mt * 16 + r + mrow][(nt0 + j) * 16 + lrow] = (__bf16)val;
      }
    }
    __syncthreads();
    // ---- phase 2: oacc += act(64x64) @ W2[chunk,:](64x256) ----
    #pragma unroll
    for (int k0 = 0; k0 < 64; k0 += 32) {
      int kb = k0 + khalf;
      Frag16 a;
      const __bf16* ap = &act[mt * 16 + lrow][kb];
      a.u[0] = *(const v4u*)ap;
      a.u[1] = *(const v4u*)(ap + 16);
      #pragma unroll
      for (int j = 0; j < 8; ++j) {
        int nc = (ntb + j) * 16 + lrow;
        const __bf16* bp = w2t + (size_t)nc * 512 + chunk * 64 + kb;
        Frag16 bb; bb.u[0] = *(const v4u*)bp; bb.u[1] = *(const v4u*)(bp + 16);
        oacc[j] = wmma_bf16(a, bb, oacc[j]);
      }
    }
    __syncthreads();
  }

  // ---- dump C-fragments to LDS (channel-major), add attn, coalesced store ----
  #pragma unroll
  for (int j = 0; j < 8; ++j) {
    #pragma unroll
    for (int r = 0; r < 8; ++r) {
      xstage[(ntb + j) * 16 + lrow][mt * 16 + r + mrow] = oacc[j][r];
    }
  }
  __syncthreads();
  float* ob = out + ((size_t)b * C_) * HW_ + hw0;
  #pragma unroll 4
  for (int i = 0; i < 64; ++i) {
    int idx = t + 256 * i; int c = idx >> 6, m = idx & 63;
    ob[(size_t)c * HW_ + m] = xstage[c][m] + attn_s[c];
  }
}

extern "C" void kernel_launch(void* const* d_in, const int* in_sizes, int n_in,
                              void* d_out, int out_size, void* d_ws, size_t ws_size,
                              hipStream_t stream) {
  (void)in_sizes; (void)n_in; (void)out_size; (void)ws_size;
  const float* x     = (const float*)d_in[0];   // cnn_features (16,256,64,64)
  const float* extra = (const float*)d_in[1];   // (16,512)
  const float* g1    = (const float*)d_in[2];
  const float* b1    = (const float*)d_in[3];
  const float* g2    = (const float*)d_in[4];
  const float* b2    = (const float*)d_in[5];
  // d_in[6] = Wq: provably unused (softmax over constant logits is uniform)
  const float* wkv   = (const float*)d_in[7];   // (512,128)
  const float* wout  = (const float*)d_in[8];   // (512,256)
  const float* W1    = (const float*)d_in[9];   // (256,1024)
  const float* W2    = (const float*)d_in[10];  // (512,256)
  float* out = (float*)d_out;

  char* ws = (char*)d_ws;
  float*  attn = (float*)ws;                        // 16*256*4   = 16 KB
  __bf16* w1t  = (__bf16*)(ws + 16384);             // 1024*256*2 = 512 KB
  __bf16* w2t  = (__bf16*)(ws + 16384 + 524288);    // 256*512*2  = 256 KB

  conv_w_kernel<<<1536, 256, 0, stream>>>(W1, W2, w1t, w2t);
  attn_row_kernel<<<B_, 256, 0, stream>>>(extra, g2, b2, wkv, wout, attn);
  ff_fused_kernel<<<B_ * 64, 256, 0, stream>>>(x, g1, b1, w1t, w2t, attn, out);
}
